// IxformerGroupQuantGemmMoE_80908593922368
// MI455X (gfx1250) — compile-verified, matrix operand order checked by reading, not run on hardware
//
#include <hip/hip_runtime.h>
#include <hip/hip_bf16.h>

typedef int v8i __attribute__((ext_vector_type(8)));
typedef int v4i_t __attribute__((vector_size(16)));   // matches async builtin proto

#define E_ 8
#define T_ 8192
#define K_ 2048
#define N_ 2048
#define BS_ 128
#define NB_ 16           // K blocks of 128

#define BM 128
#define BN 128
#define THREADS 256
#define CH 64            // K chunk per WMMA round
#define LSTR 80          // 64 data bytes + 16 pad; mult of 16 (async B128 dest align)

#if defined(__has_builtin)
#if __has_builtin(__builtin_amdgcn_global_load_async_to_lds_b128) && \
    __has_builtin(__builtin_amdgcn_s_wait_asynccnt)
#define USE_ASYNC 1
#endif
#endif

#define GPTR(p) ((__attribute__((address_space(1))) v4i_t*)(p))
#define LPTR(p) ((__attribute__((address_space(3))) v4i_t*)(p))

// Pack the low bytes of 4 int32s into one dword with 3 v_perm_b32.
__device__ __forceinline__ unsigned pack4(int4 v) {
    unsigned tlo = __builtin_amdgcn_perm((unsigned)v.y, (unsigned)v.x, 0x00000400u);
    unsigned thi = __builtin_amdgcn_perm((unsigned)v.w, (unsigned)v.z, 0x00000400u);
    return __builtin_amdgcn_perm(thi, tlo, 0x05040100u);
}

// ---------------- Pass 1: int32 -> int8 pre-pack (streamed once) ----------------
__global__ __launch_bounds__(THREADS)
void pack_kernel(const int* __restrict__ src, unsigned char* __restrict__ dst) {
    size_t idx = (size_t)blockIdx.x * THREADS + threadIdx.x;
    const int4* s = (const int4*)(src + idx * 16);
    uint4 o;
    o.x = pack4(s[0]);
    o.y = pack4(s[1]);
    o.z = pack4(s[2]);
    o.w = pack4(s[3]);
    *(uint4*)(dst + idx * 16) = o;
}

// ---------------- Pass 2: grouped w8a8 GEMM with WMMA IU8 ----------------
__global__ __launch_bounds__(THREADS)
void moe_w8a8_wmma_kernel(const unsigned char* __restrict__ A8,
                          const unsigned char* __restrict__ W8,
                          const int* __restrict__ token_count,
                          const float* __restrict__ weight_scale,
                          const float* __restrict__ input_scale,
                          const float* __restrict__ bias,
                          __hip_bfloat16* __restrict__ out)
{
    __shared__ unsigned char lA[2][BM * LSTR];   // ping-pong K=64 chunks
    __shared__ unsigned char lB[2][BN * LSTR];
    __shared__ float sIS[NB_ * BM];              // [b][row]  (transposed)
    __shared__ float sWS[NB_ * BN];              // [b][col]  (transposed)

    const int tid  = threadIdx.x;
    const int lane = tid & 31;
    const int wave = tid >> 5;
    const int hi   = lane >> 4;
    const int lo   = lane & 15;
    const int fb   = hi * 8;         // fragment byte offset for upper half-wave

    const int row_base = blockIdx.y * BM;
    const int col_base = blockIdx.x * BN;

    // Expert owning this (contiguous, BM-aligned) token tile.
    int e = 0, acc_t = 0;
    for (int i = 0; i < E_; ++i) {
        int c = token_count[i];
        if (row_base >= acc_t + c) { acc_t += c; e = i + 1; }
    }
    if (e >= E_) e = E_ - 1;

    const unsigned char* gA = A8 + (size_t)row_base * K_;
    const unsigned char* gW = W8 + (size_t)e * N_ * K_ + (size_t)col_base * K_;

    // Per-thread staging slots: 128 rows x 4 b128-chunks = 512 slots/tile, 2/thread.
    const int sr0 = (tid          ) >> 2, sc0 = (tid          ) & 3;
    const int sr1 = (tid + THREADS) >> 2, sc1 = (tid + THREADS) & 3;

#ifdef USE_ASYNC
#define ISSUE_STEP(S, BI)                                                          \
    do {                                                                           \
        size_t kb = (size_t)(S) * CH;                                              \
        __builtin_amdgcn_global_load_async_to_lds_b128(                            \
            GPTR(gA + (size_t)sr0 * K_ + kb + sc0 * 16),                           \
            LPTR(&lA[BI][sr0 * LSTR + sc0 * 16]), 0, 0);                           \
        __builtin_amdgcn_global_load_async_to_lds_b128(                            \
            GPTR(gA + (size_t)sr1 * K_ + kb + sc1 * 16),                           \
            LPTR(&lA[BI][sr1 * LSTR + sc1 * 16]), 0, 0);                           \
        __builtin_amdgcn_global_load_async_to_lds_b128(                            \
            GPTR(gW + (size_t)sr0 * K_ + kb + sc0 * 16),                           \
            LPTR(&lB[BI][sr0 * LSTR + sc0 * 16]), 0, 0);                           \
        __builtin_amdgcn_global_load_async_to_lds_b128(                            \
            GPTR(gW + (size_t)sr1 * K_ + kb + sc1 * 16),                           \
            LPTR(&lB[BI][sr1 * LSTR + sc1 * 16]), 0, 0);                           \
    } while (0)
#define WAIT_STAGE() __builtin_amdgcn_s_wait_asynccnt(0)
#else
    uint4 rA0, rA1, rW0, rW1;
#define ISSUE_STEP(S, BI)                                                          \
    do {                                                                           \
        size_t kb = (size_t)(S) * CH;                                              \
        rA0 = *(const uint4*)(gA + (size_t)sr0 * K_ + kb + sc0 * 16);              \
        rA1 = *(const uint4*)(gA + (size_t)sr1 * K_ + kb + sc1 * 16);              \
        rW0 = *(const uint4*)(gW + (size_t)sr0 * K_ + kb + sc0 * 16);              \
        rW1 = *(const uint4*)(gW + (size_t)sr1 * K_ + kb + sc1 * 16);              \
    } while (0)
#define WAIT_STAGE() do { } while (0)
#endif

    // Stage per-block dequant scales (transposed to [b][row]).
    for (int i = tid; i < NB_ * BM; i += THREADS) {
        int bb = i >> 7, r = i & (BM - 1);
        sIS[i] = input_scale[(size_t)(row_base + r) * NB_ + bb];
    }
    for (int i = tid; i < NB_ * BN; i += THREADS) {
        int bb = i >> 7, n = i & (BN - 1);
        sWS[i] = weight_scale[((size_t)e * N_ + col_base + n) * NB_ + bb];
    }

    // 8 waves in a 2x4 grid; wave tile = 64 rows x 32 cols = 4x2 WMMA tiles.
    const int wm = (wave >> 2) * 64;
    const int wn = (wave & 3) * 32;

    float facc[4][2][8];
#pragma unroll
    for (int mt = 0; mt < 4; ++mt)
#pragma unroll
        for (int nt = 0; nt < 2; ++nt)
#pragma unroll
            for (int k = 0; k < 8; ++k)
                facc[mt][nt][k] = 0.0f;

#ifndef USE_ASYNC
#define COMMIT(BI)                                                                 \
    do {                                                                           \
        *(uint4*)&lA[BI][sr0 * LSTR + sc0 * 16] = rA0;                             \
        *(uint4*)&lA[BI][sr1 * LSTR + sc1 * 16] = rA1;                             \
        *(uint4*)&lB[BI][sr0 * LSTR + sc0 * 16] = rW0;                             \
        *(uint4*)&lB[BI][sr1 * LSTR + sc1 * 16] = rW1;                             \
    } while (0)
#endif

    // Fragment loader: rows at stride LSTR, 4x b64 at {0,16,32,48}+fb.
#define LOAD_FRAG(dst, buf, rowbase)                                               \
    do {                                                                           \
        const unsigned char* p = &buf[(rowbase) * LSTR + fb];                      \
        union { v8i v; unsigned long long q[4]; } u;                               \
        u.q[0] = *(const unsigned long long*)(p +  0);                             \
        u.q[1] = *(const unsigned long long*)(p + 16);                             \
        u.q[2] = *(const unsigned long long*)(p + 32);                             \
        u.q[3] = *(const unsigned long long*)(p + 48);                             \
        dst = u.v;                                                                 \
    } while (0)

    v8i ic[4][2];

    ISSUE_STEP(0, 0);

    for (int b = 0; b < NB_; ++b) {
        // ================= chunk 0 (even step -> buf0) =================
#ifdef USE_ASYNC
        WAIT_STAGE();
        __syncthreads();                   // buf0 ready; prior buf1 readers done
        ISSUE_STEP(2 * b + 1, 1);          // fetch chunk 1 while chunk 0 computes
#else
        __syncthreads();
        COMMIT(0);
        __syncthreads();
        ISSUE_STEP(2 * b + 1, 1);
#endif
        {
            v8i af[4], bf[2];
#pragma unroll
            for (int mt = 0; mt < 4; ++mt) LOAD_FRAG(af[mt], lA[0], wm + mt * 16 + lo);
#pragma unroll
            for (int nt = 0; nt < 2; ++nt) LOAD_FRAG(bf[nt], lB[0], wn + nt * 16 + lo);
#pragma unroll
            for (int mt = 0; mt < 4; ++mt)
#pragma unroll
                for (int nt = 0; nt < 2; ++nt) {
                    v8i z = {0, 0, 0, 0, 0, 0, 0, 0};
                    ic[mt][nt] = __builtin_amdgcn_wmma_i32_16x16x64_iu8(
                            true, af[mt], true, bf[nt], z, false, false);
                }
        }

        // ================= chunk 1 (odd step -> buf1) =================
#ifdef USE_ASYNC
        WAIT_STAGE();
        __syncthreads();                   // buf1 ready; all waves done reading buf0
        if (b + 1 < NB_) ISSUE_STEP(2 * b + 2, 0);
#else
        __syncthreads();
        COMMIT(1);
        __syncthreads();
        if (b + 1 < NB_) ISSUE_STEP(2 * b + 2, 0);
#endif
        {
            // Per-lane scales; is values contiguous -> compiler merges to b64 loads.
            float wsv[2];
#pragma unroll
            for (int nt = 0; nt < 2; ++nt)
                wsv[nt] = sWS[b * BN + wn + nt * 16 + lo];
            float isv[4][8];
#pragma unroll
            for (int mt = 0; mt < 4; ++mt)
#pragma unroll
                for (int k = 0; k < 8; ++k)
                    isv[mt][k] = sIS[b * BM + wm + mt * 16 + fb + k];

            v8i af[4], bf[2];
#pragma unroll
            for (int mt = 0; mt < 4; ++mt) LOAD_FRAG(af[mt], lA[1], wm + mt * 16 + lo);
#pragma unroll
            for (int nt = 0; nt < 2; ++nt) LOAD_FRAG(bf[nt], lB[1], wn + nt * 16 + lo);
#pragma unroll
            for (int mt = 0; mt < 4; ++mt)
#pragma unroll
                for (int nt = 0; nt < 2; ++nt)
                    ic[mt][nt] = __builtin_amdgcn_wmma_i32_16x16x64_iu8(
                            true, af[mt], true, bf[nt], ic[mt][nt], false, false);

            // Dequant-FMA for this K-block.
#pragma unroll
            for (int mt = 0; mt < 4; ++mt)
#pragma unroll
                for (int nt = 0; nt < 2; ++nt)
#pragma unroll
                    for (int k = 0; k < 8; ++k)
                        facc[mt][nt][k] += (float)ic[mt][nt][k] * (isv[mt][k] * wsv[nt]);
        }
    }

    // ---- Epilogue: bias + bf16 store.
#pragma unroll
    for (int mt = 0; mt < 4; ++mt) {
#pragma unroll
        for (int nt = 0; nt < 2; ++nt) {
            int gn = col_base + wn + nt * 16 + lo;
            float bv = bias[(size_t)e * N_ + gn];
#pragma unroll
            for (int k = 0; k < 8; ++k) {
                int gm = row_base + wm + mt * 16 + k + fb;
                out[(size_t)gm * N_ + gn] = __float2bfloat16(facc[mt][nt][k] + bv);
            }
        }
    }
}

extern "C" void kernel_launch(void* const* d_in, const int* in_sizes, int n_in,
                              void* d_out, int out_size, void* d_ws, size_t ws_size,
                              hipStream_t stream) {
    (void)in_sizes; (void)n_in; (void)out_size; (void)ws_size;
    const int*   input        = (const int*)d_in[0];
    const int*   weight       = (const int*)d_in[1];
    const int*   token_count  = (const int*)d_in[2];
    const float* weight_scale = (const float*)d_in[3];
    const float* input_scale  = (const float*)d_in[4];
    const float* bias         = (const float*)d_in[5];
    __hip_bfloat16* out       = (__hip_bfloat16*)d_out;

    unsigned char* A8 = (unsigned char*)d_ws;                       // 16 MB
    unsigned char* W8 = A8 + (size_t)T_ * K_;                       // 32 MB

    // Pass 1: pre-pack int32 -> int8 (16 ints per thread).
    pack_kernel<<<dim3((T_ * (size_t)K_) / 16 / THREADS), dim3(THREADS), 0, stream>>>(input, A8);
    pack_kernel<<<dim3((E_ * (size_t)N_ * K_) / 16 / THREADS), dim3(THREADS), 0, stream>>>(weight, W8);

    // Pass 2: grouped GEMM.
    dim3 grid(N_ / BN, T_ / BM);   // (16, 64)
    dim3 block(THREADS);
    moe_w8a8_wmma_kernel<<<grid, block, 0, stream>>>(
        A8, W8, token_count, weight_scale, input_scale, bias, out);
}
